// OrthoLinear_18588618457625
// MI455X (gfx1250) — matrix operand order repaired and checked by reading, not compiled, not run on hardware
//
#include <hip/hip_runtime.h>
#include <hip/hip_bf16.h>

// ---------------------------------------------------------------------------
// Y = X @ (dequant_int4(Wpacked, scales) + CSR(ortho))^T
//   X: [8192, 4096] fp32, W: [4096, 4096], Y: [8192, 4096] fp32
// Fold sparse term into dequantized f16 weight once, then one dense f16-WMMA
// GEMM (fp32 accumulate). Compute-bound: 2.75e11 FLOP vs ~300MB HBM traffic.
// ---------------------------------------------------------------------------

#define TOKENS 8192
#define IN_F   4096
#define OUT_F  4096

typedef __attribute__((ext_vector_type(8)))  _Float16 v8h;
typedef __attribute__((ext_vector_type(16))) _Float16 v16h;
typedef __attribute__((ext_vector_type(8)))  float    v8f;
typedef __attribute__((ext_vector_type(4)))  float    v4f;
typedef __attribute__((ext_vector_type(4)))  int      v4i;

// ---- CDNA5 async global->LDS path (no data VGPRs, tracked by ASYNCcnt) -----
#if __has_builtin(__builtin_amdgcn_global_load_async_to_lds_b128) && \
    __has_builtin(__builtin_amdgcn_s_wait_asynccnt)
#define USE_ASYNC_LDS 1
#else
#define USE_ASYNC_LDS 0
#endif

#if USE_ASYNC_LDS
__device__ __forceinline__ void async_copy16(const _Float16* g, _Float16* l) {
    __builtin_amdgcn_global_load_async_to_lds_b128(
        (__attribute__((address_space(1))) v4i*)g,   // global source (AS1)
        (__attribute__((address_space(3))) v4i*)l,   // LDS destination (AS3)
        /*offset=*/0, /*cpol=*/0);
}
#endif

// -------------------------- kernel 1: X fp32 -> f16 -------------------------
__global__ __launch_bounds__(256)
void cvt_x_kernel(const float* __restrict__ x, _Float16* __restrict__ xh) {
    size_t i = ((size_t)blockIdx.x * blockDim.x + threadIdx.x) * 8;
    // X is read exactly once -> non-temporal loads keep L2 for W/Xh tiles
    v4f a = __builtin_nontemporal_load((const v4f*)(x + i));
    v4f b = __builtin_nontemporal_load((const v4f*)(x + i + 4));
    v8h h;
    h[0] = (_Float16)a[0]; h[1] = (_Float16)a[1];
    h[2] = (_Float16)a[2]; h[3] = (_Float16)a[3];
    h[4] = (_Float16)b[0]; h[5] = (_Float16)b[1];
    h[6] = (_Float16)b[2]; h[7] = (_Float16)b[3];
    *(v8h*)(xh + i) = h;
}

// ------------------ kernel 2: dequant int4 + CSR scatter -> f16 W -----------
__global__ __launch_bounds__(256)
void build_w_kernel(const int* __restrict__ bw,          // [OUT_F, IN_F/2], byte per word
                    const float* __restrict__ scales,    // [OUT_F]
                    const float* __restrict__ vals,      // [NNZ]
                    const int* __restrict__ cidx,        // [NNZ]
                    const int* __restrict__ rptr,        // [OUT_F+1]
                    _Float16* __restrict__ wh) {         // [OUT_F, IN_F] row-major
    __shared__ float row[IN_F];
    const int o = blockIdx.x;
    const int t = threadIdx.x;
    const float s = scales[o];
    const int* wrow = bw + (size_t)o * (IN_F / 2);

#pragma unroll
    for (int j = 0; j < 8; ++j) {
        int idx  = j * 256 + t;                  // coalesced word index
        int word = wrow[idx];
        int lo = ((word & 15) ^ 8) - 8;          // signed int4, low nibble = even col
        int hi = (((word >> 4) & 15) ^ 8) - 8;
        row[2 * idx]     = (float)lo * s;
        row[2 * idx + 1] = (float)hi * s;
    }
    __syncthreads();

    // ALPHA == 1.0; LDS float atomics handle duplicate column indices
    int start = rptr[o], end = rptr[o + 1];
    for (int p = start + t; p < end; p += 256)
        atomicAdd(&row[cidx[p]], vals[p]);
    __syncthreads();

#pragma unroll
    for (int j = 0; j < 16; ++j) {
        int idx = j * 256 + t;
        wh[(size_t)o * IN_F + idx] = (_Float16)row[idx];
    }
}

// ------------------------------ kernel 3: GEMM ------------------------------
// Block: 256 threads = 8 wave32. Output tile 256(M) x 128(N). K-step 32.
// Wave grid 4(M) x 2(N): each wave owns 64x64 = 4x4 tiles of 16x16.
// Double-buffered LDS (60KB); padded row stride (40 halves) keeps every
// ds_load_b128 fragment read bank-conflict-free. Tile staging uses
// GLOBAL_LOAD_ASYNC_TO_LDS_B128 (zero staging VGPRs) when available.

#define LDT 40

__device__ __forceinline__ v16h frag2(const _Float16* p0, const _Float16* p1) {
    v8h lo = *(const v8h*)p0;
    v8h hi = *(const v8h*)p1;
    return __builtin_shufflevector(lo, hi, 0,1,2,3,4,5,6,7,8,9,10,11,12,13,14,15);
}

__global__ __launch_bounds__(256, 1)
void gemm_kernel(const _Float16* __restrict__ Xh,   // [TOKENS, IN_F]
                 const _Float16* __restrict__ Wh,   // [OUT_F, IN_F]
                 float* __restrict__ out) {         // [TOKENS, OUT_F]
    __shared__ _Float16 As[2][256 * LDT];   // 40 KB
    __shared__ _Float16 Bs[2][128 * LDT];   // 20 KB

    const int K  = IN_F;
    const int n0 = blockIdx.x * 128;
    const int m0 = blockIdx.y * 256;
    const int t    = threadIdx.x;
    const int lane = t & 31;
    const int wave = t >> 5;
    const int wm = (wave >> 1) * 64;   // wave M offset (0/64/128/192)
    const int wn = (wave & 1) * 64;    // wave N offset (0/64)
    const int lr  = lane & 15;
    const int lhi = lane >> 4;

    // cooperative global->LDS: A = 1024 x 16B chunks (c=0..3), B = 512 (c=4..5)
    const int rsub = t >> 2;            // 0..63 row sub-index
    const int kc   = (t & 3) * 8;       // k-chunk in halves

    const _Float16* Ag = Xh + (size_t)m0 * K;
    const _Float16* Bg = Wh + (size_t)n0 * K;

    v8f acc[4][4] = {};

    // prologue: stage K-tile 0 into buffer 0
#if USE_ASYNC_LDS
#pragma unroll
    for (int c = 0; c < 4; ++c)
        async_copy16(Ag + (size_t)(c * 64 + rsub) * K + kc,
                     &As[0][(c * 64 + rsub) * LDT + kc]);
#pragma unroll
    for (int c = 4; c < 6; ++c)
        async_copy16(Bg + (size_t)((c - 4) * 64 + rsub) * K + kc,
                     &Bs[0][((c - 4) * 64 + rsub) * LDT + kc]);
    __builtin_amdgcn_s_wait_asynccnt(0);
#else
    {
        int4 rg[6];
#pragma unroll
        for (int c = 0; c < 4; ++c)
            rg[c] = *(const int4*)(Ag + (size_t)(c * 64 + rsub) * K + kc);
#pragma unroll
        for (int c = 4; c < 6; ++c)
            rg[c] = *(const int4*)(Bg + (size_t)((c - 4) * 64 + rsub) * K + kc);
#pragma unroll
        for (int c = 0; c < 4; ++c)
            *(int4*)(&As[0][(c * 64 + rsub) * LDT + kc]) = rg[c];
#pragma unroll
        for (int c = 4; c < 6; ++c)
            *(int4*)(&Bs[0][((c - 4) * 64 + rsub) * LDT + kc]) = rg[c];
    }
#endif
    __syncthreads();

    for (int kt = 0; kt < K; kt += 32) {
        const int cur   = (kt >> 5) & 1;
        const int nxt   = cur ^ 1;
        const bool more = (kt + 32) < K;

#if USE_ASYNC_LDS
        // issue next tile's async copies; they run behind the 16 WMMAs
        if (more) {
            const int kn = kt + 32;
#pragma unroll
            for (int c = 0; c < 4; ++c)
                async_copy16(Ag + (size_t)(c * 64 + rsub) * K + kn + kc,
                             &As[nxt][(c * 64 + rsub) * LDT + kc]);
#pragma unroll
            for (int c = 4; c < 6; ++c)
                async_copy16(Bg + (size_t)((c - 4) * 64 + rsub) * K + kn + kc,
                             &Bs[nxt][((c - 4) * 64 + rsub) * LDT + kc]);
        }
#else
        int4 rg[6];
        if (more) {
            const int kn = kt + 32;
#pragma unroll
            for (int c = 0; c < 4; ++c)
                rg[c] = *(const int4*)(Ag + (size_t)(c * 64 + rsub) * K + kn + kc);
#pragma unroll
            for (int c = 4; c < 6; ++c)
                rg[c] = *(const int4*)(Bg + (size_t)((c - 4) * 64 + rsub) * K + kn + kc);
        }
#endif

        // fragment loads per CDNA5 16-bit WMMA layouts:
        // A 16x32: lane<16 -> M=lane, K=0..7 & 16..23 ; lane>=16 -> K=8..15 & 24..31
        // B 32x16: lane<16 -> N=lane, K=0..7 & 16..23 ; lane>=16 -> K=8..15 & 24..31
        v16h af[4], bf[4];
        const _Float16* Ab = &As[cur][0];
        const _Float16* Bb = &Bs[cur][0];
#pragma unroll
        for (int tm = 0; tm < 4; ++tm) {
            const _Float16* p = Ab + (wm + tm * 16 + lr) * LDT + lhi * 8;
            af[tm] = frag2(p, p + 16);
        }
#pragma unroll
        for (int tn = 0; tn < 4; ++tn) {
            const _Float16* p = Bb + (wn + tn * 16 + lr) * LDT + lhi * 8;
            bf[tn] = frag2(p, p + 16);
        }

#pragma unroll
        for (int tm = 0; tm < 4; ++tm)
#pragma unroll
            for (int tn = 0; tn < 4; ++tn)
                acc[tm][tn] = __builtin_amdgcn_wmma_f32_16x16x32_f16(
                    false, af[tm], false, bf[tn],
                    (short)0, acc[tm][tn], false, false);

#if USE_ASYNC_LDS
        if (more) __builtin_amdgcn_s_wait_asynccnt(0);
#else
        if (more) {
#pragma unroll
            for (int c = 0; c < 4; ++c)
                *(int4*)(&As[nxt][(c * 64 + rsub) * LDT + kc]) = rg[c];
#pragma unroll
            for (int c = 4; c < 6; ++c)
                *(int4*)(&Bs[nxt][((c - 4) * 64 + rsub) * LDT + kc]) = rg[c];
        }
#endif
        __syncthreads();
    }

    // epilogue: C/D layout — VGPR v, lane l: M = v + 8*(l>>4), N = l&15.
    // Output is write-once: non-temporal stores keep Xh/Wh tiles L2-resident.
#pragma unroll
    for (int tm = 0; tm < 4; ++tm)
#pragma unroll
        for (int tn = 0; tn < 4; ++tn) {
            const int nn = n0 + wn + tn * 16 + lr;
#pragma unroll
            for (int v = 0; v < 8; ++v) {
                const int mm = m0 + wm + tm * 16 + v + 8 * lhi;
                __builtin_nontemporal_store(acc[tm][tn][v],
                                            out + (size_t)mm * OUT_F + nn);
            }
        }
}

// ------------------------------- launcher -----------------------------------
extern "C" void kernel_launch(void* const* d_in, const int* in_sizes, int n_in,
                              void* d_out, int out_size, void* d_ws, size_t ws_size,
                              hipStream_t stream) {
    const float* x       = (const float*)d_in[0];
    const int*   bw      = (const int*)d_in[1];
    const float* scales  = (const float*)d_in[2];
    const float* vals    = (const float*)d_in[3];
    const int*   cidx    = (const int*)d_in[4];
    const int*   rptr    = (const int*)d_in[5];
    float*       out     = (float*)d_out;

    // workspace: Xh (f16, 64MB) | Wh (f16, 32MB)
    _Float16* Xh = (_Float16*)d_ws;
    _Float16* Wh = (_Float16*)((char*)d_ws + (size_t)TOKENS * IN_F * sizeof(_Float16));

    {
        const size_t total = (size_t)TOKENS * IN_F;
        const int blocks = (int)(total / (256 * 8));      // 16384
        cvt_x_kernel<<<blocks, 256, 0, stream>>>(x, Xh);
    }
    build_w_kernel<<<OUT_F, 256, 0, stream>>>(bw, scales, vals, cidx, rptr, Wh);
    {
        dim3 grid(OUT_F / 128, TOKENS / 256);             // (32, 32)
        gemm_kernel<<<grid, 256, 0, stream>>>(Xh, Wh, out);
    }
}
